// E3NNModel_48421461295281
// MI455X (gfx1250) — matrix operand order repaired and checked by reading, not compiled, not run on hardware
//
#include <hip/hip_runtime.h>
#include <hip/hip_bf16.h>

// ---------------- problem constants (from reference) ----------------
#define NNODES   8192
#define NEDGES   65536
#define NGRAPHS  64
#define DIMIN    118
#define MUL0     32
#define MUL1     8
#define DIMHID   56          // MUL0 + 3*MUL1
#define DIMEDGE  64
#define NCONV    2
#define MLPIN    128         // 2*MUL0 + DIMEDGE
#define WN       1920        // total mlp output width
#define NTILES   120         // WN / 16
#define MSGW     64          // out_s(32) + out_g(8) + out_v(24)

#define INV_SQRT118  0.09205746178983233f
#define INV_SQRT128  0.08838834764831845f
#define INV_SQRT32   0.17677669529663687f
#define INV_SQRT40   0.15811388300841897f   // FAN^-0.5
#define SQRT3F       1.7320508075688772f
#define INV_SQRT3    0.5773502691896258f

typedef __attribute__((ext_vector_type(16))) _Float16 v16h;
typedef __attribute__((ext_vector_type(8)))  _Float16 v8h;
typedef __attribute__((ext_vector_type(8)))  float    v8f;

// ---------------- small helper kernels ----------------
__global__ void k_zero(float* p, int n) {
  int i = blockIdx.x * blockDim.x + threadIdx.x;
  if (i < n) p[i] = 0.0f;
}

// x[:, :32] = attrs_node @ W_emb * 118^-0.5 + b_emb ; x[:, 32:56] = 0
__global__ void k_embed(const float* __restrict__ attrs_node,
                        const float* __restrict__ W_emb,
                        const float* __restrict__ b_emb,
                        float* __restrict__ x) {
  int idx = blockIdx.x * blockDim.x + threadIdx.x;
  if (idx >= NNODES * DIMHID) return;
  int n = idx / DIMHID, c = idx % DIMHID;
  if (c < MUL0) {
    float acc = 0.0f;
    const float* an = attrs_node + (size_t)n * DIMIN;
    for (int k = 0; k < DIMIN; ++k) acc += an[k] * W_emb[k * MUL0 + c];
    x[idx] = acc * INV_SQRT118 + b_emb[c];
  } else {
    x[idx] = 0.0f;
  }
}

// Yv = sqrt(3)*normalize(rij) ; e_attr = soft_one_hot(d)
__global__ void k_edgefeat(const float* __restrict__ rijs,
                           const float* __restrict__ attrs_edge,
                           float* __restrict__ yv,
                           float* __restrict__ eattr) {
  int idx = blockIdx.x * blockDim.x + threadIdx.x;
  if (idx >= NEDGES * DIMEDGE) return;
  int e = idx >> 6, j = idx & 63;
  if (j < 3) {
    float r0 = rijs[e * 3 + 0], r1 = rijs[e * 3 + 1], r2 = rijs[e * 3 + 2];
    float inv = rsqrtf(r0 * r0 + r1 * r1 + r2 * r2);
    float c = (j == 0) ? r0 : ((j == 1) ? r1 : r2);
    yv[e * 3 + j] = SQRT3F * c * inv;
  }
  const float step = 5.0f / 65.0f;
  float center = (float)(j + 1) * step;
  float d = attrs_edge[e];
  float diff = (d - center) / step;
  eattr[(size_t)e * DIMEDGE + j] = expf(-diff * diff) * (1.0f / 1.12f);
}

// Wt[c][n][k] = f16( W_mlp[c][k][n] * 128^-0.5 )   (transpose + scale fold)
__global__ void k_wtprep(const float* __restrict__ Wm, _Float16* __restrict__ Wt) {
  int idx = blockIdx.x * blockDim.x + threadIdx.x;
  if (idx >= NCONV * WN * MLPIN) return;
  int c = idx / (WN * MLPIN);
  int rem = idx - c * (WN * MLPIN);
  int n = rem >> 7, k = rem & 127;
  Wt[idx] = (_Float16)(Wm[(size_t)c * MLPIN * WN + (size_t)k * WN + n] * INV_SQRT128);
}

// ---------------- fused edge-conv kernel (WMMA) ----------------
// One wave = 32 edges (two 16-row A-tiles sharing every B fragment -> 1
// global_load_b128 per WMMA). B fragments are register double-buffered and
// the tile loop is split into one straight-line loop per weight region
// (w1..w6), so loop bodies have no branches, the accumulator live-set stays
// small, and each region's writeback overlaps the next region's prefetch.
#define CWAVES 2
#define ETILE  32            // edges per wave
__global__ void __launch_bounds__(CWAVES * 32)
k_conv(const float* __restrict__ x,
       const _Float16* __restrict__ Wt,     // [WN][128] f16 (scaled)
       const float* __restrict__ eattr,
       const float* __restrict__ yv,
       const int* __restrict__ dst,
       const int* __restrict__ nbr,
       float* __restrict__ accum,           // [NNODES][64]
       float* __restrict__ cnt) {           // [NNODES]
  __shared__ __align__(32) _Float16 s_emb[CWAVES][ETILE][MLPIN];
  __shared__ float s_xsn[CWAVES][ETILE][MUL0];
  __shared__ float s_dot[CWAVES][ETILE][MUL1];
  __shared__ float s_yv [CWAVES][ETILE][3];
  __shared__ float s_xvn[CWAVES][ETILE][24];
  __shared__ float s_msg[CWAVES][ETILE][MSGW];
  __shared__ int   s_dste[CWAVES][ETILE];
  __shared__ int   s_nbre[CWAVES][ETILE];

  const int lane = threadIdx.x & 31;
  const int w    = threadIdx.x >> 5;
  const int tile = blockIdx.x * CWAVES + w;    // 0 .. 2047 exactly
  const int e0   = tile * ETILE;

  s_dste[w][lane] = dst[e0 + lane];
  s_nbre[w][lane] = nbr[e0 + lane];
  __syncthreads();

  // stage emb (f16), xsn, xvn, yv; zero msg accumulator
  for (int j = lane; j < ETILE * MUL0; j += 32) {
    int m = j >> 5, c = j & 31;
    float xd = x[(size_t)s_dste[w][m] * DIMHID + c];
    float xn = x[(size_t)s_nbre[w][m] * DIMHID + c];
    s_emb[w][m][c]        = (_Float16)xd;
    s_emb[w][m][MUL0 + c] = (_Float16)xn;
    s_xsn[w][m][c] = xn;
  }
  for (int j = lane; j < ETILE * 24; j += 32) {
    int m = j / 24, c = j % 24;
    s_xvn[w][m][c] = x[(size_t)s_nbre[w][m] * DIMHID + MUL0 + c];
  }
  for (int j = lane; j < ETILE * 3; j += 32) {
    int m = j / 3, c = j % 3;
    s_yv[w][m][c] = yv[(size_t)(e0 + m) * 3 + c];
  }
  for (int j = lane; j < ETILE * DIMEDGE; j += 32) {
    int m = j >> 6, c = j & 63;
    s_emb[w][m][2 * MUL0 + c] = (_Float16)eattr[(size_t)(e0 + m) * DIMEDGE + c];
    s_msg[w][m][c] = 0.0f;
  }
  __syncthreads();

  // dot[e][u] = (xvn[e][u] . Yv[e]) / sqrt(3)
  for (int j = lane; j < ETILE * MUL1; j += 32) {
    int m = j >> 3, u = j & 7;
    s_dot[w][m][u] = (s_xvn[w][m][u * 3 + 0] * s_yv[w][m][0] +
                      s_xvn[w][m][u * 3 + 1] * s_yv[w][m][1] +
                      s_xvn[w][m][u * 3 + 2] * s_yv[w][m][2]) * INV_SQRT3;
  }
  __syncthreads();

  // A fragments (two 16-row M-tiles). 16-bit A layout (matches B):
  //   row = lane%16; lanes<16 hold K {kb*32+0..15},
  //   lanes>=16 hold {kb*32+16..31}.
  const int r    = lane & 15;
  const int koff = (lane < 16) ? 0 : 16;
  v16h afr0[4], afr1[4];
#pragma unroll
  for (int kb = 0; kb < 4; ++kb) {
    union { v16h v; v8h h[2]; } au;
    au.h[0] = *(const v8h*)&s_emb[w][r][kb * 32 + koff];
    au.h[1] = *(const v8h*)&s_emb[w][r][kb * 32 + 8 + koff];
    afr0[kb] = au.v;
    au.h[0] = *(const v8h*)&s_emb[w][16 + r][kb * 32 + koff];
    au.h[1] = *(const v8h*)&s_emb[w][16 + r][kb * 32 + 8 + koff];
    afr1[kb] = au.v;
  }

  const int ncol  = lane & 15;
  const int koffB = (lane < 16) ? 0 : 16;   // B: lanes<16 K=0..15, lanes>=16 K=16..31
  const int hi8   = (lane < 16) ? 0 : 8;    // C/D: M = v + hi8
  const int wc8   = ncol & 7;

  // helper macros (all names in scope at expansion sites)
#define LOADB(DST, T)                                                           \
  { const _Float16* wp_ = Wt + (size_t)((T) * 16 + ncol) * MLPIN + koffB;       \
    _Pragma("unroll")                                                           \
    for (int kb = 0; kb < 4; ++kb) DST[kb] = *(const v16h*)(wp_ + kb * 32); }

#define WMMA2(C0, C1, B)                                                        \
  { _Pragma("unroll")                                                           \
    for (int kb = 0; kb < 4; ++kb)                                              \
      C0 = __builtin_amdgcn_wmma_f32_16x16x32_f16(                              \
               false, afr0[kb], false, B[kb], (short)0, C0, false, false);      \
    _Pragma("unroll")                                                           \
    for (int kb = 0; kb < 4; ++kb)                                              \
      C1 = __builtin_amdgcn_wmma_f32_16x16x32_f16(                              \
               false, afr1[kb], false, B[kb], (short)0, C1, false, false); }

#define LOADL(L0, L1, ARR, U)                                                   \
  { _Pragma("unroll")                                                           \
    for (int v = 0; v < 8; ++v) {                                               \
      L0[v] = ARR[w][v + hi8][(U)];                                             \
      L1[v] = ARR[w][16 + v + hi8][(U)];                                        \
    } }

  v16h b0[4], b1[4];
  LOADB(b0, 0);

  // ---- region w1 [0,64) + w2 [64,80): out_s ----
  {
    v8f as0_0 = {}, as1_0 = {}, as0_1 = {}, as1_1 = {};
    for (int tp = 0; tp < 64; tp += 2) {       // w1: lhs = xsn[u], u = tp>>1
      LOADB(b1, tp + 1);
      const int u = tp >> 1;
      v8f l0, l1;  LOADL(l0, l1, s_xsn, u);
      { v8f c0 = {}, c1 = {}; WMMA2(c0, c1, b0);
        as0_0 += c0 * l0; as0_1 += c1 * l1; }
      LOADB(b0, tp + 2);                       // tp=62 prefetches tile 64 (w2)
      { v8f c0 = {}, c1 = {}; WMMA2(c0, c1, b1);
        as1_0 += c0 * l0; as1_1 += c1 * l1; }
    }
    for (int tp = 64; tp < 80; tp += 2) {      // w2: lhs = dot[u]
      LOADB(b1, tp + 1);
      const int u = (tp - 64) >> 1;
      v8f l0, l1;  LOADL(l0, l1, s_dot, u);
      { v8f c0 = {}, c1 = {}; WMMA2(c0, c1, b0);
        as0_0 += c0 * l0; as0_1 += c1 * l1; }
      LOADB(b0, tp + 2);                       // tp=78 prefetches tile 80 (w3)
      { v8f c0 = {}, c1 = {}; WMMA2(c0, c1, b1);
        as1_0 += c0 * l0; as1_1 += c1 * l1; }
    }
    // writeback out_s (lane-unique plain stores), overlaps w3 prefetch
#pragma unroll
    for (int v = 0; v < 8; ++v) {
      s_msg[w][v + hi8][ncol]            = as0_0[v];
      s_msg[w][v + hi8][ncol + 16]       = as1_0[v];
      s_msg[w][16 + v + hi8][ncol]       = as0_1[v];
      s_msg[w][16 + v + hi8][ncol + 16]  = as1_1[v];
    }
  }

  // ---- region w3 [80,96) + w4 [96,100): out_g ----
  {
    v8f ag_0 = {}, ag_1 = {};
    for (int tp = 80; tp < 96; tp += 2) {      // w3: lhs = xsn[u]
      LOADB(b1, tp + 1);
      const int u = ((tp - 80) << 1) + (ncol >> 3);
      v8f l0, l1;  LOADL(l0, l1, s_xsn, u);
      { v8f c0 = {}, c1 = {}; WMMA2(c0, c1, b0);
        ag_0 += c0 * l0; ag_1 += c1 * l1; }
      LOADB(b0, tp + 2);
      LOADL(l0, l1, s_xsn, u + 2);
      { v8f c0 = {}, c1 = {}; WMMA2(c0, c1, b1);
        ag_0 += c0 * l0; ag_1 += c1 * l1; }
    }
    for (int tp = 96; tp < 100; tp += 2) {     // w4: lhs = dot[u]
      LOADB(b1, tp + 1);
      const int u = ((tp - 96) << 1) + (ncol >> 3);
      v8f l0, l1;  LOADL(l0, l1, s_dot, u);
      { v8f c0 = {}, c1 = {}; WMMA2(c0, c1, b0);
        ag_0 += c0 * l0; ag_1 += c1 * l1; }
      LOADB(b0, tp + 2);
      LOADL(l0, l1, s_dot, u + 2);
      { v8f c0 = {}, c1 = {}; WMMA2(c0, c1, b1);
        ag_0 += c0 * l0; ag_1 += c1 * l1; }
    }
    // writeback out_g (combine lane pairs ncol / ncol+8)
#pragma unroll
    for (int v = 0; v < 8; ++v) {
      atomicAdd(&s_msg[w][v + hi8][MUL0 + wc8],      ag_0[v]);
      atomicAdd(&s_msg[w][16 + v + hi8][MUL0 + wc8], ag_1[v]);
    }
  }

  // ---- region w5 [100,116): out_v += xsn[u]*W (Yv applied at writeback) ----
  {
    v8f av_0 = {}, av_1 = {};
    for (int tp = 100; tp < 116; tp += 2) {
      LOADB(b1, tp + 1);
      const int u = ((tp - 100) << 1) + (ncol >> 3);
      v8f l0, l1;  LOADL(l0, l1, s_xsn, u);
      { v8f c0 = {}, c1 = {}; WMMA2(c0, c1, b0);
        av_0 += c0 * l0; av_1 += c1 * l1; }
      LOADB(b0, tp + 2);
      LOADL(l0, l1, s_xsn, u + 2);
      { v8f c0 = {}, c1 = {}; WMMA2(c0, c1, b1);
        av_0 += c0 * l0; av_1 += c1 * l1; }
    }
#pragma unroll
    for (int v = 0; v < 8; ++v) {
      int m = v + hi8;
      atomicAdd(&s_msg[w][m][40 + wc8 * 3 + 0], av_0[v] * s_yv[w][m][0]);
      atomicAdd(&s_msg[w][m][40 + wc8 * 3 + 1], av_0[v] * s_yv[w][m][1]);
      atomicAdd(&s_msg[w][m][40 + wc8 * 3 + 2], av_0[v] * s_yv[w][m][2]);
      m = 16 + v + hi8;
      atomicAdd(&s_msg[w][m][40 + wc8 * 3 + 0], av_1[v] * s_yv[w][m][0]);
      atomicAdd(&s_msg[w][m][40 + wc8 * 3 + 1], av_1[v] * s_yv[w][m][1]);
      atomicAdd(&s_msg[w][m][40 + wc8 * 3 + 2], av_1[v] * s_yv[w][m][2]);
    }
  }

  // ---- region w6 [116,120): out_v[i] += xvn[u][i]*W (direct LDS atomics) ----
#define W6TILE(C0, C1, U)                                                       \
  { _Pragma("unroll")                                                           \
    for (int v = 0; v < 8; ++v) {                                               \
      int m = v + hi8;                                                          \
      const float* p = &s_xvn[w][m][(U) * 3];                                   \
      atomicAdd(&s_msg[w][m][40 + wc8 * 3 + 0], (C0)[v] * p[0]);                \
      atomicAdd(&s_msg[w][m][40 + wc8 * 3 + 1], (C0)[v] * p[1]);                \
      atomicAdd(&s_msg[w][m][40 + wc8 * 3 + 2], (C0)[v] * p[2]);                \
      m = 16 + v + hi8;                                                         \
      p = &s_xvn[w][m][(U) * 3];                                                \
      atomicAdd(&s_msg[w][m][40 + wc8 * 3 + 0], (C1)[v] * p[0]);                \
      atomicAdd(&s_msg[w][m][40 + wc8 * 3 + 1], (C1)[v] * p[1]);                \
      atomicAdd(&s_msg[w][m][40 + wc8 * 3 + 2], (C1)[v] * p[2]);                \
    } }
  for (int tp = 116; tp < 120; tp += 2) {
    LOADB(b1, tp + 1);
    const int u = ((tp - 116) << 1) + (ncol >> 3);
    { v8f c0 = {}, c1 = {}; WMMA2(c0, c1, b0);
      W6TILE(c0, c1, u); }
    if (tp + 2 < 120) LOADB(b0, tp + 2);
    { v8f c0 = {}, c1 = {}; WMMA2(c0, c1, b1);
      W6TILE(c0, c1, u + 2); }
  }
#undef W6TILE
#undef LOADL
#undef WMMA2
#undef LOADB
  __syncthreads();

  // scatter: msg * FAN^-0.5 into node accumulators
  for (int j = lane; j < ETILE * MSGW; j += 32) {
    int m = j >> 6, c = j & 63;
    atomicAdd(&accum[(size_t)s_dste[w][m] * MSGW + c], s_msg[w][m][c] * INV_SQRT40);
  }
  atomicAdd(&cnt[s_dste[w][lane]], 1.0f);
}

// node update: mean, relu, gate, residual
__global__ void k_update(const float* __restrict__ x,
                         const float* __restrict__ accum,
                         const float* __restrict__ cnt,
                         float* __restrict__ xnew) {
  int idx = blockIdx.x * blockDim.x + threadIdx.x;
  if (idx >= NNODES * DIMHID) return;
  int n = idx / DIMHID, c = idx % DIMHID;
  float cn = fmaxf(cnt[n], 1.0f);
  const float* a = accum + (size_t)n * MSGW;
  float out;
  if (c < MUL0) {
    out = x[idx] + fmaxf(a[c] / cn, 0.0f);
  } else {
    int k = c - MUL0;
    int u = k / 3;
    float g = fmaxf(a[MUL0 + u] / cn, 0.0f);
    float v = a[40 + k] / cn;
    out = x[idx] + v * g;
  }
  xnew[idx] = out;
}

// graph readout: per-node MLP head then segment mean
__global__ void k_readout(const float* __restrict__ x,
                          const float* __restrict__ W_bg0,
                          const float* __restrict__ b_bg0,
                          const float* __restrict__ W_pg,
                          const float* __restrict__ b_pg,
                          const int* __restrict__ batch,
                          float* __restrict__ gsum,
                          float* __restrict__ gcnt) {
  int n = blockIdx.x * blockDim.x + threadIdx.x;
  if (n >= NNODES) return;
  const float* xn = x + (size_t)n * DIMHID;
  float acc = 0.0f;
  for (int j = 0; j < MUL0; ++j) {
    float bg = 0.0f;
    for (int u = 0; u < MUL0; ++u) bg += xn[u] * W_bg0[u * 40 + j];
    bg = bg * INV_SQRT32 + b_bg0[j];
    acc += fmaxf(bg, 0.0f) * W_pg[j];
  }
  float out = acc * INV_SQRT32 + b_pg[0];
  int g = batch[n];
  atomicAdd(&gsum[g], out);
  atomicAdd(&gcnt[g], 1.0f);
}

__global__ void k_final(const float* __restrict__ gsum,
                        const float* __restrict__ gcnt,
                        float* __restrict__ out) {
  int g = threadIdx.x;
  if (g < NGRAPHS) out[g] = gsum[g] / fmaxf(gcnt[g], 1.0f);
}

// ---------------- host-side launch ----------------
extern "C" void kernel_launch(void* const* d_in, const int* in_sizes, int n_in,
                              void* d_out, int out_size, void* d_ws, size_t ws_size,
                              hipStream_t stream) {
  const float* attrs_node = (const float*)d_in[0];
  const float* rijs       = (const float*)d_in[1];
  const float* attrs_edge = (const float*)d_in[2];
  const float* W_emb      = (const float*)d_in[3];
  const float* b_emb      = (const float*)d_in[4];
  const float* W_mlp      = (const float*)d_in[5];
  const float* W_bg0      = (const float*)d_in[6];
  const float* b_bg0      = (const float*)d_in[7];
  // d_in[8] = W_bg1 (unused in forward)
  const float* W_pg       = (const float*)d_in[9];
  const float* b_pg       = (const float*)d_in[10];
  const int*   idxe       = (const int*)d_in[11];
  const int*   batch      = (const int*)d_in[12];
  const int* dst = idxe;
  const int* nbr = idxe + NEDGES;

  // workspace carve (all chunks are 256-byte multiples)
  char* ws = (char*)d_ws;
  float* xA    = (float*)ws;  ws += (size_t)NNODES * DIMHID * 4;
  float* xB    = (float*)ws;  ws += (size_t)NNODES * DIMHID * 4;
  float* eattr = (float*)ws;  ws += (size_t)NEDGES * DIMEDGE * 4;
  float* yv    = (float*)ws;  ws += (size_t)NEDGES * 3 * 4;
  float* accum = (float*)ws;  ws += (size_t)NNODES * MSGW * 4;
  float* cnt   = (float*)ws;  ws += (size_t)NNODES * 4;
  float* gsum  = (float*)ws;  ws += (size_t)NGRAPHS * 4;
  float* gcnt  = (float*)ws;  ws += (size_t)NGRAPHS * 4;
  _Float16* Wt = (_Float16*)ws;  // NCONV * WN * 128 f16

  // weight prep (transpose + f16 + fold MLP_IN^-0.5)
  {
    int n = NCONV * WN * MLPIN;
    k_wtprep<<<(n + 255) / 256, 256, 0, stream>>>(W_mlp, Wt);
  }
  // node embedding
  {
    int n = NNODES * DIMHID;
    k_embed<<<(n + 255) / 256, 256, 0, stream>>>(attrs_node, W_emb, b_emb, xA);
  }
  // edge features
  {
    int n = NEDGES * DIMEDGE;
    k_edgefeat<<<(n + 255) / 256, 256, 0, stream>>>(rijs, attrs_edge, yv, eattr);
  }

  float* xcur = xA;
  float* xnxt = xB;
  for (int c = 0; c < NCONV; ++c) {
    // zero accum + cnt (contiguous: NNODES*64 + NNODES)
    {
      int n = NNODES * (MSGW + 1);
      k_zero<<<(n + 255) / 256, 256, 0, stream>>>(accum, n);
    }
    k_conv<<<NEDGES / (ETILE * CWAVES), CWAVES * 32, 0, stream>>>(
        xcur, Wt + (size_t)c * WN * MLPIN, eattr, yv, dst, nbr, accum, cnt);
    {
      int n = NNODES * DIMHID;
      k_update<<<(n + 255) / 256, 256, 0, stream>>>(xcur, accum, cnt, xnxt);
    }
    float* tmp = xcur; xcur = xnxt; xnxt = tmp;
  }

  // readout
  {
    int n = 2 * NGRAPHS;   // gsum + gcnt contiguous
    k_zero<<<1, 256, 0, stream>>>(gsum, n);
  }
  k_readout<<<NNODES / 64, 64, 0, stream>>>(xcur, W_bg0, b_bg0, W_pg, b_pg,
                                            batch, gsum, gcnt);
  k_final<<<1, 64, 0, stream>>>(gsum, gcnt, (float*)d_out);
}
